// PairPotentials_50903952392739
// MI455X (gfx1250) — matrix operand order repaired and checked by reading, not compiled, not run on hardware
//
#include <hip/hip_runtime.h>
#include <hip/hip_bf16.h>
#include <math.h>
#include <stdint.h>

typedef __attribute__((ext_vector_type(16))) _Float16 v16h;
typedef __attribute__((ext_vector_type(8)))  float    v8f;

#define NMAX      2048
#define HID       32
#define CUT2      6.25f
#define BLOCK     256
#define ROWS_PER_BLOCK 16
#define CAP       4096   // compacted-distance queue capacity per block (expected ~450), multiple of 16

__device__ __forceinline__ float fast_tanh(float x) {
#if __has_builtin(__builtin_amdgcn_tanhf)
    return __builtin_amdgcn_tanhf(x);   // V_TANH_F32 on CDNA5
#else
    return tanhf(x);
#endif
}

__device__ __forceinline__ void wait_asynccnt0() {
#if __has_builtin(__builtin_amdgcn_s_wait_asynccnt)
    __builtin_amdgcn_s_wait_asynccnt(0);
#else
    asm volatile("s_wait_asynccnt 0" ::: "memory");
#endif
}

__global__ __launch_bounds__(BLOCK)
void PairPotentials_50903952392739_kernel(const float* __restrict__ xyz,
                                          const float* __restrict__ cell,
                                          const float* __restrict__ W1,
                                          const float* __restrict__ b1,
                                          const float* __restrict__ W2,
                                          const float* __restrict__ b2,
                                          float* __restrict__ out,
                                          int N) {
    __shared__ float s_xyz[NMAX * 3];
    __shared__ float s_dist[CAP];
    __shared__ unsigned s_count;
    __shared__ float s_sum;

    // ---- stage coordinates into LDS via async copy (ASYNCcnt-tracked) ----
    {
        const unsigned bytes   = (unsigned)(N * 3) * 4u;
        const unsigned aligned = bytes & ~15u;
        const unsigned long long base = (unsigned long long)(uintptr_t)xyz;
        for (unsigned b = (unsigned)threadIdx.x * 16u; b < aligned; b += BLOCK * 16u) {
            const unsigned lds_addr = (unsigned)(uintptr_t)((char*)s_xyz + b);
            asm volatile("global_load_async_to_lds_b128 %0, %1, %2"
                         :: "v"(lds_addr), "v"(b), "s"(base) : "memory");
        }
        // scalar tail (not taken when 12*N is 16B-aligned, as here)
        for (unsigned t = aligned / 4u + threadIdx.x; t < (unsigned)(N * 3); t += BLOCK)
            s_xyz[t] = xyz[t];
        wait_asynccnt0();
    }
    if (threadIdx.x == 0) { s_count = 0u; s_sum = 0.0f; }
    __syncthreads();

    const float Lx = cell[0], Ly = cell[1], Lz = cell[2];
    const float hx = 0.5f * Lx, hy = 0.5f * Ly, hz = 0.5f * Lz;

    // ---- phase 1: all-pairs distance + mask, compact into LDS queue ----
    const int i0 = blockIdx.x * ROWS_PER_BLOCK;
    for (int ii = 0; ii < ROWS_PER_BLOCK; ++ii) {
        const int i = i0 + ii;
        const float xi = s_xyz[i * 3 + 0];
        const float yi = s_xyz[i * 3 + 1];
        const float zi = s_xyz[i * 3 + 2];
        for (int j = threadIdx.x; j < N; j += BLOCK) {
            if (j <= i) continue;            // upper triangle only
            float dx = s_xyz[j * 3 + 0] - xi;
            float dy = s_xyz[j * 3 + 1] - yi;
            float dz = s_xyz[j * 3 + 2] - zi;
            // minimum image convention (matches reference offsets)
            dx += (dx < -hx ? Lx : 0.0f) - (dx >= hx ? Lx : 0.0f);
            dy += (dy < -hy ? Ly : 0.0f) - (dy >= hy ? Ly : 0.0f);
            dz += (dz < -hz ? Lz : 0.0f) - (dz >= hz ? Lz : 0.0f);
            const float r2 = dx * dx + dy * dy + dz * dz;
            if (r2 < CUT2 && r2 > 0.0f) {
                unsigned idx = atomicAdd(&s_count, 1u);   // ds_add_rtn_u32
                if (idx < CAP) s_dist[idx] = sqrtf(r2);
            }
        }
    }
    __syncthreads();

    const unsigned total = s_count;
    const int used   = (int)(total < CAP ? total : CAP);
    const int padded = (used + 15) & ~15;

    // zero-pad the queue so phase 2 loads are unconditional (no exec branch)
    for (int t = used + (int)threadIdx.x; t < padded; t += BLOCK) s_dist[t] = 0.0f;
    __syncthreads();

    // ---- phase 2: MLP over compacted pairs via WMMA ----
    const int lane = threadIdx.x & 31;
    const int wave = threadIdx.x >> 5;
    const int m    = lane & 15;                 // A-matrix row this lane feeds
    const int kb   = (lane < 16) ? 0 : 8;       // K-group base per ISA A layout

    // per-lane slice of W1/b1 matching the 16-bit 16x32 A-matrix VGPR layout:
    // slot s<8  -> K = kb + s        (VGPRs 0..3)
    // slot s>=8 -> K = 16 + kb + s-8 (VGPRs 4..7)
    float w1r[16], b1r[16];
#pragma unroll
    for (int s = 0; s < 16; ++s) {
        const int k = (s < 8) ? (kb + s) : (16 + kb + (s - 8));
        w1r[s] = W1[k];
        b1r[s] = b1[k];
    }

    // B matrix 32x16 f16: column 0 = W2, all other columns zero.
    // lane 0 holds col0 K=0..7,16..23, lane 16 holds col0 K=8..15,24..31.
    v16h bmat;
#pragma unroll
    for (int s = 0; s < 16; ++s) bmat[s] = (_Float16)0.0f;
    if (lane == 0 || lane == 16) {
        const int kb2 = (lane == 0) ? 0 : 8;
#pragma unroll
        for (int s = 0; s < 16; ++s) bmat[s] = (_Float16)W2[(s < 8) ? (kb2 + s) : (16 + kb2 + (s - 8))];
    }

    v8f acc = {};
    const int nch = padded >> 4;
    for (int c = wave; c < nch; c += (BLOCK / 32)) {   // wave-uniform loop
        const int idx = (c << 4) + m;
        const bool valid = idx < used;
        const float d = s_dist[idx];                   // unconditional ds_load
        v16h a;
#pragma unroll
        for (int s = 0; s < 16; ++s) {
            const float h = valid ? fast_tanh(fmaf(d, w1r[s], b1r[s])) : 0.0f;
            a[s] = (_Float16)h;
        }
        acc = __builtin_amdgcn_wmma_f32_16x16x32_f16(
            false, a, false, bmat, (short)0, acc, false, false);
    }

    // ---- reduce: all non-col0 entries of D are exactly 0 (B zero there) ----
    float p = acc[0] + acc[1] + acc[2] + acc[3] + acc[4] + acc[5] + acc[6] + acc[7];
#pragma unroll
    for (int off = 16; off > 0; off >>= 1) p += __shfl_xor(p, off, 32);
    if (lane == 0) atomicAdd(&s_sum, p);        // ds_add_f32
    __syncthreads();
    if (threadIdx.x == 0)
        atomicAdd(out, s_sum + (float)used * b2[0]);   // global_atomic_add_f32
}

extern "C" void kernel_launch(void* const* d_in, const int* in_sizes, int n_in,
                              void* d_out, int out_size, void* d_ws, size_t ws_size,
                              hipStream_t stream) {
    const float* xyz  = (const float*)d_in[0];
    const float* cell = (const float*)d_in[1];
    const float* W1   = (const float*)d_in[2];
    const float* b1   = (const float*)d_in[3];
    const float* W2   = (const float*)d_in[4];
    const float* b2   = (const float*)d_in[5];
    const int N = in_sizes[0] / 3;

    hipMemsetAsync(d_out, 0, sizeof(float), stream);

    dim3 grid((N + ROWS_PER_BLOCK - 1) / ROWS_PER_BLOCK);
    dim3 block(BLOCK);
    PairPotentials_50903952392739_kernel<<<grid, block, 0, stream>>>(
        xyz, cell, W1, b1, W2, b2, (float*)d_out, N);
}
